// CrossAttention_22668837388609
// MI455X (gfx1250) — matrix operand order repaired and checked by reading, not compiled, not run on hardware
//
#include <hip/hip_runtime.h>
#include <hip/hip_bf16.h>
#include <stdint.h>

// ---------------------------------------------------------------------------
// CrossAttention for MI455X (gfx1250, wave32, WMMA bf16)
//   q = meme @ Wq^T + bq ; k = text @ Wk^T + bk ; v = emoji @ Wv^T + bv
//   out = softmax(q k^T) v            (no scale, no mask, f32 out)
// Pipeline:
//   1) proj_kernel x3 : f32 GEMM via bf16 WMMA, writes Qbf,Kbf (row major)
//                       and Vt (transposed [b][a][l]) as bf16
//   2) scores_kernel  : S[b,i,j] = Q.K^T  (bf16 WMMA, f32 out)
//   3) softmax_kernel : row softmax of S, writes bf16 P aliased into S rows
//   4) out_kernel     : O = P @ V  (bf16 WMMA, f32 out -> d_out)
// ---------------------------------------------------------------------------

#define HID  1024
#define ATT  1024
#define LEN  1024
#define NBAT 16

typedef __bf16 v16bf __attribute__((ext_vector_type(16)));
typedef __bf16 v8bf  __attribute__((ext_vector_type(8)));
typedef __bf16 v4bf  __attribute__((ext_vector_type(4)));
typedef float  v8f   __attribute__((ext_vector_type(8)));

// fp32 -> bf16 round-to-nearest-even (bit manipulation; no __bf16 cast needed)
static __device__ __forceinline__ __bf16 f2bf(float f) {
  uint32_t u = __builtin_bit_cast(uint32_t, f);
  u += 0x7FFFu + ((u >> 16) & 1u);
  uint16_t h = (uint16_t)(u >> 16);
  return __builtin_bit_cast(__bf16, h);
}

static __device__ __forceinline__ v16bf pack16(v8bf lo, v8bf hi) {
  return __builtin_shufflevector(lo, hi, 0,1,2,3,4,5,6,7,8,9,10,11,12,13,14,15);
}

#define LDS_PITCH 40  // 32 halves + 8 pad -> 80B rows: conflict-free, 16B aligned

// ---------------------------------------------------------------------------
// Kernel 1: projection GEMM  out[m,n] = sum_h X[m,h]*W[n,h] + bias[n]
// block = 256 thr (8 waves), tile 128x128, K-step 32, both operands via LDS
// transpose_out==0 -> out is bf16 [B*L][ATT];  ==1 -> Vt bf16 [b][ATT][LEN]
// ---------------------------------------------------------------------------
__global__ __launch_bounds__(256) void proj_kernel(
    const float* __restrict__ X, const float* __restrict__ W,
    const float* __restrict__ bias, __bf16* __restrict__ out,
    int transpose_out)
{
  __shared__ __bf16 Xs[128 * LDS_PITCH];
  __shared__ __bf16 Ws[128 * LDS_PITCH];

  const int t    = threadIdx.x;
  const int lane = t & 31, wave = t >> 5;
  const int ln   = lane & 15, hi16 = lane >> 4;
  const int m0   = blockIdx.y * 128;   // row in flattened [B*L]
  const int n0   = blockIdx.x * 128;   // col in [ATT]

  v8f acc[8] = {};

  for (int k0 = 0; k0 < HID; k0 += 32) {
    // cooperative load + f32->bf16 convert of X and W tiles [128 x 32]
#pragma unroll
    for (int it = 0; it < 4; ++it) {
      int idx = it * 256 + t;            // 1024 float4 slots = 128 rows * 8
      int r = idx >> 3, c4 = idx & 7;
      float4 xv = *(const float4*)(X + (size_t)(m0 + r) * HID + k0 + c4 * 4);
      v4bf xb = { f2bf(xv.x), f2bf(xv.y), f2bf(xv.z), f2bf(xv.w) };
      *(v4bf*)&Xs[r * LDS_PITCH + c4 * 4] = xb;
      float4 wv = *(const float4*)(W + (size_t)(n0 + r) * HID + k0 + c4 * 4);
      v4bf wb = { f2bf(wv.x), f2bf(wv.y), f2bf(wv.z), f2bf(wv.w) };
      *(v4bf*)&Ws[r * LDS_PITCH + c4 * 4] = wb;
    }
    __syncthreads();

    // A fragment: wave's 16 rows; lane holds row, K runs {0..7,16..23}(+8 hi)
    const __bf16* ap = &Xs[(wave * 16 + ln) * LDS_PITCH + hi16 * 8];
    v16bf afrag = pack16(*(const v8bf*)ap, *(const v8bf*)(ap + 16));

#pragma unroll
    for (int s = 0; s < 8; ++s) {
      const __bf16* bp = &Ws[(s * 16 + ln) * LDS_PITCH + hi16 * 8];
      v16bf bfrag = pack16(*(const v8bf*)bp, *(const v8bf*)(bp + 16));
      acc[s] = __builtin_amdgcn_wmma_f32_16x16x32_bf16(
          false, afrag, false, bfrag, (short)0, acc[s], false, false);
    }
    __syncthreads();
  }

  // epilogue: bias + bf16 store
#pragma unroll
  for (int s = 0; s < 8; ++s) {
    int a = n0 + s * 16 + ln;
    float bv = bias[a];
    if (!transpose_out) {
#pragma unroll
      for (int r = 0; r < 8; ++r) {
        int m = m0 + hi16 * 8 + r;
        out[(size_t)m * ATT + a] = f2bf(acc[s][r] + bv);
      }
    } else {
      int b = m0 >> 10;                       // / LEN (block within one batch)
      int l = (m0 & (LEN - 1)) + hi16 * 8;
      __bf16* dst = out + ((size_t)b * ATT + a) * LEN + l;  // contiguous in l
#pragma unroll
      for (int r = 0; r < 8; ++r) dst[r] = f2bf(acc[s][r] + bv);
    }
  }
}

// ---------------------------------------------------------------------------
// Kernel 2: scores  S[b,i,j] = sum_a Q[b,i,a]*K[b,j,a]
// K tile (shared across waves) staged in LDS; Q fragments direct from L2.
// ---------------------------------------------------------------------------
__global__ __launch_bounds__(256) void scores_kernel(
    const __bf16* __restrict__ Q, const __bf16* __restrict__ Kmat,
    float* __restrict__ S)
{
  __shared__ __bf16 Ks[128 * LDS_PITCH];
  const int t = threadIdx.x, lane = t & 31, wave = t >> 5;
  const int ln = lane & 15, hi16 = lane >> 4;
  const int b  = blockIdx.z;
  const int i0 = blockIdx.y * 128;
  const int j0 = blockIdx.x * 128;
  const __bf16* Qb = Q    + (size_t)b * LEN * ATT;
  const __bf16* Kb = Kmat + (size_t)b * LEN * ATT;

  v8f acc[8] = {};

  for (int k0 = 0; k0 < ATT; k0 += 32) {
#pragma unroll
    for (int it = 0; it < 2; ++it) {
      int idx = it * 256 + t;            // 512 chunks of 8 halves = 128 x 32
      int r = idx >> 2, c8 = idx & 3;
      v8bf v = *(const v8bf*)(Kb + (size_t)(j0 + r) * ATT + k0 + c8 * 8);
      *(v8bf*)&Ks[r * LDS_PITCH + c8 * 8] = v;
    }
    __syncthreads();

    const __bf16* ap = Qb + (size_t)(i0 + wave * 16 + ln) * ATT + k0 + hi16 * 8;
    v16bf afrag = pack16(*(const v8bf*)ap, *(const v8bf*)(ap + 16));

#pragma unroll
    for (int s = 0; s < 8; ++s) {
      const __bf16* bp = &Ks[(s * 16 + ln) * LDS_PITCH + hi16 * 8];
      v16bf bfrag = pack16(*(const v8bf*)bp, *(const v8bf*)(bp + 16));
      acc[s] = __builtin_amdgcn_wmma_f32_16x16x32_bf16(
          false, afrag, false, bfrag, (short)0, acc[s], false, false);
    }
    __syncthreads();
  }

#pragma unroll
  for (int s = 0; s < 8; ++s) {
    int j = j0 + s * 16 + ln;
#pragma unroll
    for (int r = 0; r < 8; ++r) {
      int i = i0 + wave * 16 + hi16 * 8 + r;
      S[((size_t)b * LEN + i) * LEN + j] = acc[s][r];
    }
  }
}

// ---------------------------------------------------------------------------
// Kernel 3: row softmax. One 256-thread block per row (4 f32 per thread).
// Writes bf16 probabilities aliased into the same row (first L*2 bytes);
// safe: all reads complete (in regs) before any write (barriers between).
// ---------------------------------------------------------------------------
__global__ __launch_bounds__(256) void softmax_kernel(float* __restrict__ S)
{
  __shared__ float red[256];
  const int t = threadIdx.x;
  float* rp = S + (size_t)blockIdx.x * LEN;

  float4 v = *(const float4*)(rp + t * 4);
  float mx = fmaxf(fmaxf(v.x, v.y), fmaxf(v.z, v.w));
  red[t] = mx; __syncthreads();
  for (int off = 128; off > 0; off >>= 1) {
    if (t < off) red[t] = fmaxf(red[t], red[t + off]);
    __syncthreads();
  }
  mx = red[0]; __syncthreads();

  float e0 = __expf(v.x - mx), e1 = __expf(v.y - mx);
  float e2 = __expf(v.z - mx), e3 = __expf(v.w - mx);
  red[t] = e0 + e1 + e2 + e3; __syncthreads();
  for (int off = 128; off > 0; off >>= 1) {
    if (t < off) red[t] += red[t + off];
    __syncthreads();
  }
  float inv = 1.0f / red[0];

  __bf16* pp = (__bf16*)rp;                       // alias row start
  v4bf o = { f2bf(e0 * inv), f2bf(e1 * inv), f2bf(e2 * inv), f2bf(e3 * inv) };
  *(v4bf*)(pp + t * 4) = o;
}

// ---------------------------------------------------------------------------
// Kernel 4: O[b,i,a] = sum_j P[b,i,j] * V[b,j,a], V given transposed [b][a][j]
// P rows are bf16 with pitch LEN*2 halves (aliased over f32 S rows).
// ---------------------------------------------------------------------------
__global__ __launch_bounds__(256) void out_kernel(
    const float* __restrict__ Sbase, const __bf16* __restrict__ Vt,
    float* __restrict__ O)
{
  __shared__ __bf16 Vs[128 * LDS_PITCH];
  const int t = threadIdx.x, lane = t & 31, wave = t >> 5;
  const int ln = lane & 15, hi16 = lane >> 4;
  const int b  = blockIdx.z;
  const int i0 = blockIdx.y * 128;
  const int a0 = blockIdx.x * 128;
  const __bf16* Vb = Vt + (size_t)b * ATT * LEN;   // [a][j], contiguous in j

  v8f acc[8] = {};

  for (int k0 = 0; k0 < LEN; k0 += 32) {           // k = j
#pragma unroll
    for (int it = 0; it < 2; ++it) {
      int idx = it * 256 + t;
      int r = idx >> 2, c8 = idx & 3;
      v8bf v = *(const v8bf*)(Vb + (size_t)(a0 + r) * LEN + k0 + c8 * 8);
      *(v8bf*)&Vs[r * LDS_PITCH + c8 * 8] = v;
    }
    __syncthreads();

    int i = i0 + wave * 16 + ln;
    const __bf16* prow = (const __bf16*)(Sbase + ((size_t)b * LEN + i) * LEN);
    const __bf16* ap = prow + k0 + hi16 * 8;
    v16bf afrag = pack16(*(const v8bf*)ap, *(const v8bf*)(ap + 16));

#pragma unroll
    for (int s = 0; s < 8; ++s) {
      const __bf16* bp = &Vs[(s * 16 + ln) * LDS_PITCH + hi16 * 8];
      v16bf bfrag = pack16(*(const v8bf*)bp, *(const v8bf*)(bp + 16));
      acc[s] = __builtin_amdgcn_wmma_f32_16x16x32_bf16(
          false, afrag, false, bfrag, (short)0, acc[s], false, false);
    }
    __syncthreads();
  }

#pragma unroll
  for (int s = 0; s < 8; ++s) {
    int a = a0 + s * 16 + ln;
#pragma unroll
    for (int r = 0; r < 8; ++r) {
      int i = i0 + wave * 16 + hi16 * 8 + r;
      O[((size_t)b * LEN + i) * ATT + a] = acc[s][r];
    }
  }
}

// ---------------------------------------------------------------------------
extern "C" void kernel_launch(void* const* d_in, const int* in_sizes, int n_in,
                              void* d_out, int out_size, void* d_ws, size_t ws_size,
                              hipStream_t stream) {
  (void)in_sizes; (void)n_in; (void)out_size; (void)ws_size;
  const float* meme  = (const float*)d_in[0];
  const float* text  = (const float*)d_in[1];
  const float* emoji = (const float*)d_in[2];
  const float* Wq = (const float*)d_in[3];
  const float* bq = (const float*)d_in[4];
  const float* Wk = (const float*)d_in[5];
  const float* bk = (const float*)d_in[6];
  const float* Wv = (const float*)d_in[7];
  const float* bv = (const float*)d_in[8];
  float* out = (float*)d_out;

  // workspace: Qbf(32MB) | Kbf(32MB) | Vt(32MB) | S(64MB, bf16 P aliased)
  const size_t QKV_BYTES = (size_t)NBAT * LEN * ATT * 2;  // 32 MB each
  char* ws = (char*)d_ws;
  __bf16* Qb = (__bf16*)(ws);
  __bf16* Kb = (__bf16*)(ws + QKV_BYTES);
  __bf16* Vt = (__bf16*)(ws + 2 * QKV_BYTES);
  float*  S  = (float*) (ws + 3 * QKV_BYTES);

  dim3 blk(256);
  dim3 gProj(ATT / 128, (NBAT * LEN) / 128);          // 8 x 128
  proj_kernel<<<gProj, blk, 0, stream>>>(meme,  Wq, bq, Qb, 0);
  proj_kernel<<<gProj, blk, 0, stream>>>(text,  Wk, bk, Kb, 0);
  proj_kernel<<<gProj, blk, 0, stream>>>(emoji, Wv, bv, Vt, 1);

  dim3 gAttn(LEN / 128, LEN / 128, NBAT);             // 8 x 8 x 16
  scores_kernel<<<gAttn, blk, 0, stream>>>(Qb, Kb, S);

  softmax_kernel<<<dim3(NBAT * LEN), blk, 0, stream>>>(S);

  dim3 gOut(ATT / 128, LEN / 128, NBAT);
  out_kernel<<<gOut, blk, 0, stream>>>(S, Vt, out);
}
